// BackProjNet_43198781063629
// MI455X (gfx1250) — matrix operand order repaired and checked by reading, not compiled, not run on hardware
//
#include <hip/hip_runtime.h>
#include <math.h>

// ---------------------------------------------------------------------------
// BackProjNet for MI455X (gfx1250, wave32, WMMA).
//   Kernel 1 (bpn_prep):  pre-swizzle fc2 weights into WMMA A-fragment order.
//   Kernel 2 (bpn_fused): per (b,view) slice: conv1(8->176,K3)+GELU in fp32
//                         -> f16 transposed h in LDS -> conv2(176->88,K3) as
//                         WMMA f32<=f16 GEMM (M=88,N=512,K=528 padded 576),
//                         2 N-tiles per iteration sharing each A fragment,
//                         flat rolled tile loop so LDS B-loads cannot be
//                         LICM-hoisted into register spills.
//   Kernel 3 (bpn_gather): 4-tap interpolation gather -> final output.
// ---------------------------------------------------------------------------

typedef __attribute__((ext_vector_type(16))) _Float16 v16h;
typedef __attribute__((ext_vector_type(2)))  _Float16 v2h;
typedef __attribute__((ext_vector_type(8)))  float    v8f;

#define B_      2
#define C_      8
#define V_      360
#define D_      512
#define R_      11
#define COUT1   176                    // hidden channels
#define COUT2   88                     // R_*C_
#define LTOT    (V_*D_*R_)             // 2,027,520
#define NIDX    (128*128*360)          // 5,898,240

// conv2 GEMM tiling (WMMA 16x16x32 f16)
#define KSTEPS  18                     // K = 3 * 192 / 32 (cin 176 padded to 192)
#define MT      6                      // ceil(88/16)
#define NT      32                     // 512/16
#define HT_LD   192                    // padded cin leading dim of hT
#define HT_ROWS 514                    // d = -1..512 halo rows
#define X_LD    520                    // x row stride (floats)

#define AFRAG_HALFS (MT*KSTEPS*32*16)  // 55,296 halves
#define AFRAG_BYTES (AFRAG_HALFS*2)    // 110,592 B (multiple of 256)
#define HT_BYTES    (HT_ROWS*HT_LD*2)  // 197,376 B
#define X_BYTES     (C_*X_LD*4)        // 16,640 B
#define SMEM_BYTES  (HT_BYTES + X_BYTES) // 214,016 B  (< 320 KB LDS/WGP)

// ---------------------------------------------------------------------------
// Kernel 1: build WMMA A fragments for conv2 weights.
// A tile is 16x32 f16 (M x K). In this code's contiguous k order (B matches), element e of lane L holds
//   M = L&15,  K = e + 16*(L>>4)
// Global K index kappa = k*192 + cin (k = conv tap, cin padded 176->192).
// Padded rows (co>=88) and padded cin (>=176) are zero.
// ---------------------------------------------------------------------------
extern "C" __global__ __launch_bounds__(256)
void bpn_prep(const float* __restrict__ fc2_w, _Float16* __restrict__ afrag)
{
    int tid = blockIdx.x * 256 + threadIdx.x;
    if (tid >= AFRAG_HALFS) return;
    int e     = tid & 15;
    int lane  = (tid >> 4) & 31;
    int s     = (tid >> 9) % KSTEPS;          // k-step 0..17
    int mtile = tid / (KSTEPS * 32 * 16);
    int k     = s / 6;                        // conv tap (0..2), uniform per step
    int kidx  = e + ((lane >> 4) << 4);
    int cin   = ((s - k * 6) << 5) + kidx;    // 0..191
    int co    = mtile * 16 + (lane & 15);
    float val = (co < COUT2 && cin < COUT1)
                    ? fc2_w[(co * COUT1 + cin) * 3 + k] : 0.0f;
    afrag[tid] = (_Float16)val;
}

// ---------------------------------------------------------------------------
// Kernel 2: fused MLP per slice. One workgroup (8 wave32) per (b, v).
// ---------------------------------------------------------------------------
extern "C" __global__ __launch_bounds__(256)
void bpn_fused(const float* __restrict__ input,
               const float* __restrict__ fc1_w,
               const float* __restrict__ fc1_b,
               const float* __restrict__ fc2_b,
               const _Float16* __restrict__ afrag,
               float* __restrict__ outbuf)
{
    extern __shared__ __align__(32) char smem[];
    _Float16* hT = (_Float16*)smem;                 // [HT_ROWS][HT_LD] f16
    float*    xs = (float*)(smem + HT_BYTES);       // [8][X_LD] f32 (halo col)

    const int t    = threadIdx.x;
    const int wv   = t >> 5;        // wave id 0..7
    const int lane = t & 31;
    const int hi   = lane >> 4;     // half-wave select
    const int ln   = lane & 15;
    const int sid  = blockIdx.x;
    const int b    = sid / V_;
    const int v    = sid - b * V_;

    // --- zero hT: covers halo rows (d=-1,512) and cin pad (176..191) -------
    uint4 z4 = {0u, 0u, 0u, 0u};
    for (int i = t; i < HT_BYTES / 16; i += 256)
        ((uint4*)hT)[i] = z4;

    // --- stage input slice x[8][512] into LDS with zero halo columns ------
    if (t < 16) { int ci = t >> 1; xs[ci * X_LD + ((t & 1) ? 513 : 0)] = 0.0f; }
    const float* xin = input + ((size_t)(b * C_) * V_ + v) * D_;
    for (int e = t; e < C_ * (D_ / 4); e += 256) {
        int ci = e >> 7;                  // / 128
        int d4 = (e & 127) << 2;
        float4 val = *(const float4*)(xin + (size_t)ci * ((size_t)V_ * D_) + d4);
        float* dst = &xs[ci * X_LD + 1 + d4];   // halo offset -> scalar stores
        dst[0] = val.x; dst[1] = val.y; dst[2] = val.z; dst[3] = val.w;
    }
    __syncthreads();

    // --- conv1 (8->176, K=3, pad 1) + exact GELU -> hT (f16, transposed) --
    // Wave wv owns rows d = wv, wv+8, ... : x reads are uniform-address LDS
    // broadcasts; each lane produces a co-pair and stores packed 2xf16 (b32,
    // 4 B lane stride -> conflict-free). fc1 weights live in registers.
    for (int j = 0; j < 3; ++j) {
        int co0 = (lane << 1) + (j << 6);          // 0..190, step 2 per lane
        if (co0 < COUT1) {
            float wr[48];                          // taps for co0 and co0+1
            #pragma unroll
            for (int q = 0; q < 12; ++q) {
                float4 w4 = *(const float4*)(fc1_w + co0 * 24 + q * 4);
                wr[q*4+0] = w4.x; wr[q*4+1] = w4.y;
                wr[q*4+2] = w4.z; wr[q*4+3] = w4.w;
            }
            float bb0 = fc1_b[co0], bb1 = fc1_b[co0 + 1];
            for (int d = wv; d < D_; d += 8) {
                float a0 = bb0, a1 = bb1;
                #pragma unroll
                for (int ci = 0; ci < 8; ++ci) {
                    const float* xr = &xs[ci * X_LD + d];  // uniform address
                    float x0 = xr[0], x1 = xr[1], x2 = xr[2];
                    a0 = fmaf(wr[ci*3+0],      x0, a0);
                    a0 = fmaf(wr[ci*3+1],      x1, a0);
                    a0 = fmaf(wr[ci*3+2],      x2, a0);
                    a1 = fmaf(wr[24+ci*3+0],   x0, a1);
                    a1 = fmaf(wr[24+ci*3+1],   x1, a1);
                    a1 = fmaf(wr[24+ci*3+2],   x2, a1);
                }
                float g0 = 0.5f * a0 * (1.0f + erff(a0 * 0.70710678118654752f));
                float g1 = 0.5f * a1 * (1.0f + erff(a1 * 0.70710678118654752f));
                v2h pack; pack.x = (_Float16)g0; pack.y = (_Float16)g1;
                *(v2h*)(hT + (d + 1) * HT_LD + co0) = pack;   // ds_store_b32
            }
        }
    }
    __syncthreads();

    // --- conv2 as WMMA GEMM: Y[88][512] = W2(88x576) x B(576x512) ---------
    // Wave wv owns d block [wv*64, wv*64+64) = 4 N-tiles. Flat rolled loop:
    // mtile varies fastest (new A fragment each iteration), npair = it/6
    // selects the N-tile pair, so B addresses depend on the induction
    // variable (not loop-invariant -> no LICM hoist -> no spills). Two
    // independent WMMA chains share each A fragment.
    const v16h* abase = (const v16h*)afrag;
    const int dblk = wv << 6;                       // wave's d base

    #pragma clang loop unroll(disable)
    for (int it = 0; it < 2 * MT; ++it) {
        int mtile = (it >= MT) ? (it - MT) : it;    // it % 6, fastest
        int npair = (it >= MT) ? 1 : 0;             // N-tile pair select
        int nb    = dblk + (npair << 5);            // d base of this pair

        v8f acc0 = {}, acc1 = {};
        const v16h* ap = abase + (size_t)(mtile * KSTEPS) * 32 + lane;

        #pragma unroll
        for (int s = 0; s < KSTEPS; ++s) {
            int k    = s / 6;                       // uniform conv tap
            int cin0 = (s - k * 6) << 5;            // cin base of this k-step
            v16h a = ap[s * 32];                    // 2x global_load_b128
            // B fragment: lane ln holds column d; elems 0..15 are
            // K = 16*hi .. 16*hi+15 -> contiguous 32B in transposed hT.
            const _Float16* hb = hT + cin0 + (hi << 4);
            int r0 = nb + ln + k;
            v16h b0 = *(const v16h*)(hb + (size_t)(r0     ) * HT_LD);
            v16h b1 = *(const v16h*)(hb + (size_t)(r0 + 16) * HT_LD);
            acc0 = __builtin_amdgcn_wmma_f32_16x16x32_f16(
                       false, a, false, b0, (short)0, acc0, false, false);
            acc1 = __builtin_amdgcn_wmma_f32_16x16x32_f16(
                       false, a, false, b1, (short)0, acc1, false, false);
        }

        // --- bias + permuted store: co=c*11+r ; l=(v*512+d)*11+r ----------
        #pragma unroll
        for (int j = 0; j < 8; ++j) {
            int m  = j + (hi << 3);                 // C/D fragment row map
            int co = mtile * 16 + m;
            if (co < COUT2) {
                int c = co / R_;
                int r = co - c * R_;
                float bias = fc2_b[co];
                size_t base = (size_t)(b * C_ + c) * LTOT
                            + (size_t)v * (D_ * R_) + r;
                int d0 = nb + ln;
                outbuf[base + (size_t)(d0     ) * R_] = acc0[j] + bias;
                outbuf[base + (size_t)(d0 + 16) * R_] = acc1[j] + bias;
            }
        }
    }
}

// ---------------------------------------------------------------------------
// Kernel 3: back-projection interpolation gather.
// ---------------------------------------------------------------------------
extern "C" __global__ __launch_bounds__(256)
void bpn_gather(const float* __restrict__ indices,
                const float* __restrict__ outbuf,
                float* __restrict__ dout)
{
    int i = blockIdx.x * 256 + threadIdx.x;
    if (i >= NIDX) return;
    __builtin_prefetch(indices + i + 8192, 0, 0);   // global_prefetch_b8

    float idx = indices[i];
    float il  = floorf(idx);
    float w   = idx - il;                 // frac (indices >= 0)
    float fw  = floorf(w * 5.0f);
    float low_w    = w * 5.0f - fw;
    float low_ind  = il * 11.0f + 5.0f + fw;
    float high_ind = low_ind + 6.0f;
    if (high_ind >= (float)(LTOT - 1)) high_ind = (float)(LTOT - 2);
    int li  = (int)low_ind;
    int hix = (int)high_ind;
    float omlw = 1.0f - low_w;
    float omw  = 1.0f - w;

    #pragma unroll
    for (int bc = 0; bc < B_ * C_; ++bc) {
        const float* p = outbuf + (size_t)bc * LTOT;
        float lo = p[li]  * omlw + p[li  + 1] * low_w;
        float hq = p[hix] * omlw + p[hix + 1] * low_w;
        dout[(size_t)bc * NIDX + i] = lo * omw + hq * w;
    }
}

// ---------------------------------------------------------------------------
// Host launcher. Workspace: [A-fragments 110,592 B][out buffer ~123.8 MB f32]
// ---------------------------------------------------------------------------
extern "C" void kernel_launch(void* const* d_in, const int* in_sizes, int n_in,
                              void* d_out, int out_size, void* d_ws, size_t ws_size,
                              hipStream_t stream)
{
    const float* input   = (const float*)d_in[0];
    const float* indices = (const float*)d_in[1];
    const float* fc1_w   = (const float*)d_in[2];
    const float* fc1_b   = (const float*)d_in[3];
    const float* fc2_w   = (const float*)d_in[4];
    const float* fc2_b   = (const float*)d_in[5];
    float* dout = (float*)d_out;

    _Float16* afrag = (_Float16*)d_ws;
    float* outbuf   = (float*)((char*)d_ws + AFRAG_BYTES);

    bpn_prep<<<AFRAG_HALFS / 256, 256, 0, stream>>>(fc2_w, afrag);
    bpn_fused<<<B_ * V_, 256, SMEM_BYTES, stream>>>(input, fc1_w, fc1_b, fc2_b,
                                                    afrag, outbuf);
    bpn_gather<<<(NIDX + 255) / 256, 256, 0, stream>>>(indices, outbuf, dout);
}
